// LSTMDecoder_83889301225892
// MI455X (gfx1250) — compile-verified, matrix operand order checked by reading, not run on hardware
//
#include <hip/hip_runtime.h>
#include <math.h>

// ---------------- problem constants (match reference) ----------------
#define B_   64
#define T_   512
#define I_   64
#define H_   512
#define C_   512
#define NA_  24
#define G4_  2048   // 4*H

// ---------------- bf16 helpers / vector types ----------------
typedef __bf16 bf16;
typedef __bf16 v8bf  __attribute__((ext_vector_type(8)));
typedef __bf16 v16bf __attribute__((ext_vector_type(16)));
typedef float  v8f   __attribute__((ext_vector_type(8)));

__device__ __forceinline__ bf16 f2bf(float f) {
  union { float f; unsigned u; } v; v.f = f;
  unsigned r = v.u + 0x7FFFu + ((v.u >> 16) & 1u);   // round-to-nearest-even
  unsigned short h = (unsigned short)(r >> 16);
  bf16 o; __builtin_memcpy(&o, &h, 2); return o;
}
__device__ __forceinline__ float bf2f(bf16 b) {
  unsigned short h; __builtin_memcpy(&h, &b, 2);
  union { unsigned u; float f; } v; v.u = ((unsigned)h) << 16; return v.f;
}
__device__ __forceinline__ float sigf(float x) { return 1.f / (1.f + __expf(-x)); }

// Load a 16-bit WMMA A/B fragment for one K=32 slab.
// p points at (row base)+khalf already; elements {0..7} from +k, {8..15} from +k+16.
__device__ __forceinline__ v16bf load_frag(const bf16* __restrict__ p, int k) {
  v8bf lo = *(const v8bf*)(p + k);
  v8bf hi = *(const v8bf*)(p + k + 16);
  v16bf r;
#pragma unroll
  for (int i = 0; i < 8; ++i) { r[i] = lo[i]; r[i + 8] = hi[i]; }
  return r;
}
__device__ __forceinline__ v8f wmma_bf16(v16bf a, v16bf b, v8f c) {
  return __builtin_amdgcn_wmma_f32_16x16x32_bf16(false, a, false, b, (short)0, c, false, false);
}

__device__ __forceinline__ void storeOut(float* p, float v) { *p = v; }
__device__ __forceinline__ void storeOut(bf16*  p, float v) { *p = f2bf(v); }

// ---------------- WMMA GEMM, 16x16 tile/wave: out[M,N] = A[M,K] @ W[N,K]^T + bias --------
// Grid: (N/64, M/16), block 128 (4 waves). Used for the small latency-bound GEMMs.
template <typename OutT>
__global__ void gemm_bias_kernel(const bf16* __restrict__ A, long lda,
                                 const bf16* __restrict__ W,
                                 const float* __restrict__ bias,
                                 OutT* __restrict__ out, long ldo, int K) {
  const int lane  = threadIdx.x & 31;
  const int wave  = threadIdx.x >> 5;
  const int n0    = (blockIdx.x * 4 + wave) * 16;
  const int m0    = blockIdx.y * 16;
  const int rlo   = lane & 15;
  const int khalf = (lane >> 4) * 8;

  const bf16* aRow = A + (long)(m0 + rlo) * lda + khalf;
  const bf16* wRow = W + (long)(n0 + rlo) * K  + khalf;

  v8f acc = {};
  for (int k = 0; k < K; k += 32) {
    if (k + 64 < K) __builtin_prefetch(wRow + k + 64, 0, 0);   // global_prefetch_b8
    acc = wmma_bf16(load_frag(aRow, k), load_frag(wRow, k), acc);
  }
  const float bn    = bias ? bias[n0 + rlo] : 0.f;
  const int   rbase = (lane >> 4) * 8;           // C layout: lanes>=16 hold rows 8..15
#pragma unroll
  for (int r = 0; r < 8; ++r) {
    long row = m0 + rbase + r;
    storeOut(&out[row * ldo + n0 + rlo], acc[r] + bn);
  }
}

// ---------------- WMMA GEMM, 16x64 tile/wave (A-fragment reuse x4) ----------------
// Grid: (N/256, M/16), block 128 (4 waves). Used for the big parallel input GEMMs.
template <typename OutT>
__global__ void gemm_bias_nb4_kernel(const bf16* __restrict__ A, long lda,
                                     const bf16* __restrict__ W,
                                     const float* __restrict__ bias,
                                     OutT* __restrict__ out, long ldo, int K) {
  const int lane  = threadIdx.x & 31;
  const int wave  = threadIdx.x >> 5;
  const int n0    = (blockIdx.x * 4 + wave) * 64;   // 64 output cols per wave
  const int m0    = blockIdx.y * 16;
  const int rlo   = lane & 15;
  const int khalf = (lane >> 4) * 8;

  const bf16* aRow = A + (long)(m0 + rlo) * lda + khalf;
  const bf16* w0   = W + (long)(n0 + rlo) * K + khalf;   // subtile rows differ by 16*K

  v8f acc0 = {}, acc1 = {}, acc2 = {}, acc3 = {};
  for (int k = 0; k < K; k += 32) {
    v16bf a  = load_frag(aRow, k);                 // loaded once, used 4x
    v16bf b0 = load_frag(w0 + 0L * 16 * K, k);
    v16bf b1 = load_frag(w0 + 1L * 16 * K, k);
    v16bf b2 = load_frag(w0 + 2L * 16 * K, k);
    v16bf b3 = load_frag(w0 + 3L * 16 * K, k);
    acc0 = wmma_bf16(a, b0, acc0);
    acc1 = wmma_bf16(a, b1, acc1);
    acc2 = wmma_bf16(a, b2, acc2);
    acc3 = wmma_bf16(a, b3, acc3);
  }
  const int rbase = (lane >> 4) * 8;
#pragma unroll
  for (int j = 0; j < 4; ++j) {
    v8f acc = (j == 0) ? acc0 : (j == 1) ? acc1 : (j == 2) ? acc2 : acc3;
    const int nc = n0 + j * 16 + rlo;
    const float bn = bias ? bias[nc] : 0.f;
#pragma unroll
    for (int r = 0; r < 8; ++r) {
      long row = m0 + rbase + r;
      storeOut(&out[row * ldo + nc], acc[r] + bn);
    }
  }
}

// ---------------- fused recurrent-GEMM + LSTM cell step ----------------
// gates = gx + hPrev @ Whh^T ; cell update ; write h (bf16 + optional fp32).
// Grid: (H/16, B/16), block 128; wave index == gate index (i,f,g,o).
__global__ void lstm_step_kernel(const bf16* __restrict__ gx,    long gxStride,
                                 const bf16* __restrict__ hPrev, long hPrevStride,
                                 const bf16* __restrict__ Whh,   // [4H, H] row-major
                                 float* __restrict__ c,          // [B, H]
                                 bf16*  __restrict__ hOut,  long hOutStride,
                                 float* __restrict__ hOutF, long hOutFStride,
                                 int H) {
  __shared__ float g4[4][16][16];
  const int lane  = threadIdx.x & 31;
  const int wave  = threadIdx.x >> 5;       // gate: 0=i 1=f 2=g 3=o
  const int u0    = blockIdx.x * 16;
  const int m0    = blockIdx.y * 16;
  const int rlo   = lane & 15;
  const int khalf = (lane >> 4) * 8;

  const bf16* aRow = hPrev + (long)(m0 + rlo) * hPrevStride + khalf;
  const bf16* wRow = Whh   + (long)(wave * H + u0 + rlo) * H + khalf;

  v8f acc = {};
  for (int k = 0; k < H; k += 32)
    acc = wmma_bf16(load_frag(aRow, k), load_frag(wRow, k), acc);

  const int rbase = (lane >> 4) * 8;
#pragma unroll
  for (int r = 0; r < 8; ++r) {
    int m = rbase + r;
    float v = acc[r] + bf2f(gx[(long)(m0 + m) * gxStride + (long)wave * H + u0 + rlo]);
    g4[wave][m][rlo] = v;
  }
  __syncthreads();

  for (int idx = threadIdx.x; idx < 256; idx += 128) {
    int m = idx >> 4, u = idx & 15;
    float gi = g4[0][m][u], gf = g4[1][m][u], gg = g4[2][m][u], go = g4[3][m][u];
    long b = m0 + m, unit = u0 + u;
    float cold = c[b * H + unit];
    float cn = sigf(gf) * cold + sigf(gi) * tanhf(gg);
    float h  = sigf(go) * tanhf(cn);
    c[b * H + unit] = cn;
    hOut[b * hOutStride + unit] = f2bf(h);
    if (hOutF) hOutF[b * hOutFStride + unit] = h;
  }
}

// ---------------- attention / elementwise helpers (fp32 VALU, small cost) ----------------
__global__ void attn_scores_kernel(const float* __restrict__ enc,        // [B,T,H]
                                   const float* __restrict__ hq, long hqStride,
                                   float* __restrict__ scores, int T, int H) {
  int gw   = blockIdx.x * (blockDim.x >> 5) + (threadIdx.x >> 5);
  int lane = threadIdx.x & 31;
  int b = gw / T, t = gw % T;
  const float* e = enc + ((long)b * T + t) * H;
  const float* q = hq + (long)b * hqStride;
  float s = 0.f;
  for (int h = lane; h < H; h += 32) s += e[h] * q[h];
#pragma unroll
  for (int off = 16; off; off >>= 1) s += __shfl_xor(s, off, 32);
  if (lane == 0) scores[(long)b * T + t] = s;
}

__global__ void softmax_kernel(const float* __restrict__ scores,
                               float* __restrict__ aw, int T) {
  __shared__ float red[256];
  int b = blockIdx.x;
  const float* s = scores + (long)b * T;
  float m = -1e30f;
  for (int t = threadIdx.x; t < T; t += 256) m = fmaxf(m, s[t]);
  red[threadIdx.x] = m; __syncthreads();
  for (int w = 128; w; w >>= 1) {
    if (threadIdx.x < w) red[threadIdx.x] = fmaxf(red[threadIdx.x], red[threadIdx.x + w]);
    __syncthreads();
  }
  m = red[0]; __syncthreads();
  float sum = 0.f;
  for (int t = threadIdx.x; t < T; t += 256) sum += __expf(s[t] - m);
  red[threadIdx.x] = sum; __syncthreads();
  for (int w = 128; w; w >>= 1) {
    if (threadIdx.x < w) red[threadIdx.x] += red[threadIdx.x + w];
    __syncthreads();
  }
  float inv = 1.f / red[0];
  for (int t = threadIdx.x; t < T; t += 256) aw[(long)b * T + t] = __expf(s[t] - m) * inv;
}

__global__ void attn_ctx_kernel(const float* __restrict__ enc, const float* __restrict__ aw,
                                float* __restrict__ ctx, int T, int H) {
  int b = blockIdx.y;
  int h = blockIdx.x * 256 + threadIdx.x;
  const float* e = enc + (long)b * T * H + h;
  const float* a = aw + (long)b * T;
  float s = 0.f;
  for (int t = 0; t < T; ++t) s += a[t] * e[(long)t * H];
  ctx[(long)b * H + h] = s;
}

__global__ void xi_kernel(const float* __restrict__ ctx,
                          const float* __restrict__ predPrev, long predStride,
                          bf16* __restrict__ xi, int H) {
  long i = (long)blockIdx.x * 256 + threadIdx.x;
  long b = i / H, h = i % H;
  float v = ctx[i] + (predPrev ? predPrev[b * predStride + h] : 0.f);
  xi[i] = f2bf(v);
}

__global__ void cvt_kernel(const float* __restrict__ in, bf16* __restrict__ out, long n) {
  long i = (long)blockIdx.x * 256 + threadIdx.x;
  if (i < n) out[i] = f2bf(in[i]);
}
__global__ void addbias_kernel(const float* a, const float* b, float* o, int n) {
  int i = blockIdx.x * 256 + threadIdx.x;
  if (i < n) o[i] = a[i] + b[i];
}
__global__ void zero_f_kernel(float* p, long n) {
  long i = (long)blockIdx.x * 256 + threadIdx.x; if (i < n) p[i] = 0.f;
}
__global__ void zero_bf_kernel(bf16* p, long n) {
  long i = (long)blockIdx.x * 256 + threadIdx.x; if (i < n) p[i] = f2bf(0.f);
}

// ---------------- host orchestration ----------------
extern "C" void kernel_launch(void* const* d_in, const int* in_sizes, int n_in,
                              void* d_out, int out_size, void* d_ws, size_t ws_size,
                              hipStream_t stream) {
  (void)in_sizes; (void)n_in; (void)out_size; (void)ws_size;

  const float* x      = (const float*)d_in[0];
  const float* w_ih0  = (const float*)d_in[1];
  const float* w_hh0  = (const float*)d_in[2];
  const float* b_ih0  = (const float*)d_in[3];
  const float* b_hh0  = (const float*)d_in[4];
  const float* w_ih1  = (const float*)d_in[5];
  const float* w_hh1  = (const float*)d_in[6];
  const float* b_ih1  = (const float*)d_in[7];
  const float* b_hh1  = (const float*)d_in[8];
  const float* w_ihd  = (const float*)d_in[9];
  const float* w_hhd  = (const float*)d_in[10];
  const float* b_ihd  = (const float*)d_in[11];
  const float* b_hhd  = (const float*)d_in[12];
  const float* w_fc   = (const float*)d_in[13];
  const float* b_fc   = (const float*)d_in[14];
  float* out = (float*)d_out;

  char* wsb = (char*)d_ws;
  size_t off = 0;
  auto alloc = [&](size_t bytes) -> void* {
    off = (off + 255) & ~(size_t)255;
    void* p = wsb + off; off += bytes; return p;
  };

  bf16*  xbf    = (bf16*) alloc((size_t)B_ * T_ * I_ * 2);
  bf16*  wih0b  = (bf16*) alloc((size_t)G4_ * I_ * 2);
  bf16*  whh0b  = (bf16*) alloc((size_t)G4_ * H_ * 2);
  bf16*  wih1b  = (bf16*) alloc((size_t)G4_ * H_ * 2);
  bf16*  whh1b  = (bf16*) alloc((size_t)G4_ * H_ * 2);
  bf16*  wihdb  = (bf16*) alloc((size_t)G4_ * H_ * 2);
  bf16*  whhdb  = (bf16*) alloc((size_t)G4_ * H_ * 2);
  bf16*  wfcb   = (bf16*) alloc((size_t)C_ * H_ * 2);
  float* bias0  = (float*)alloc(G4_ * 4);
  float* bias1  = (float*)alloc(G4_ * 4);
  float* biasd  = (float*)alloc(G4_ * 4);
  bf16*  Gx0    = (bf16*) alloc((size_t)B_ * T_ * G4_ * 2);   // [B,T,4H]
  bf16*  Gx1    = (bf16*) alloc((size_t)B_ * T_ * G4_ * 2);
  bf16*  H0     = (bf16*) alloc((size_t)B_ * T_ * H_ * 2);    // [B,T,H]
  float* enc    = (float*)alloc((size_t)B_ * T_ * H_ * 4);    // [B,T,H]
  bf16*  h1p    = (bf16*) alloc((size_t)2 * B_ * H_ * 2);     // ping-pong
  float* c0     = (float*)alloc((size_t)B_ * H_ * 4);
  float* c1     = (float*)alloc((size_t)B_ * H_ * 4);
  bf16*  zerob  = (bf16*) alloc((size_t)B_ * H_ * 2);
  float* scores = (float*)alloc((size_t)B_ * T_ * 4);
  float* aw     = (float*)alloc((size_t)B_ * T_ * 4);
  float* ctx    = (float*)alloc((size_t)B_ * H_ * 4);
  bf16*  xib    = (bf16*) alloc((size_t)B_ * H_ * 2);
  bf16*  hdecb  = (bf16*) alloc((size_t)B_ * H_ * 2);
  float* hdecf  = (float*)alloc((size_t)B_ * H_ * 4);
  bf16*  Gxd    = (bf16*) alloc((size_t)B_ * G4_ * 2);

  auto cvt = [&](const float* in, bf16* o, long n) {
    cvt_kernel<<<dim3((unsigned)((n + 255) / 256)), 256, 0, stream>>>(in, o, n);
  };
  cvt(x,     xbf,   (long)B_ * T_ * I_);
  cvt(w_ih0, wih0b, (long)G4_ * I_);
  cvt(w_hh0, whh0b, (long)G4_ * H_);
  cvt(w_ih1, wih1b, (long)G4_ * H_);
  cvt(w_hh1, whh1b, (long)G4_ * H_);
  cvt(w_ihd, wihdb, (long)G4_ * H_);
  cvt(w_hhd, whhdb, (long)G4_ * H_);
  cvt(w_fc,  wfcb,  (long)C_ * H_);
  addbias_kernel<<<G4_ / 256, 256, 0, stream>>>(b_ih0, b_hh0, bias0, G4_);
  addbias_kernel<<<G4_ / 256, 256, 0, stream>>>(b_ih1, b_hh1, bias1, G4_);
  addbias_kernel<<<G4_ / 256, 256, 0, stream>>>(b_ihd, b_hhd, biasd, G4_);
  zero_f_kernel <<<(B_ * H_ + 255) / 256, 256, 0, stream>>>(c0, (long)B_ * H_);
  zero_f_kernel <<<(B_ * H_ + 255) / 256, 256, 0, stream>>>(c1, (long)B_ * H_);
  zero_bf_kernel<<<(B_ * H_ + 255) / 256, 256, 0, stream>>>(zerob, (long)B_ * H_);

  const dim3 stepGrid(H_ / 16, B_ / 16);          // 32 x 4
  const dim3 bigGrid(G4_ / 256, (B_ * T_) / 16);  // 8 x 2048, 16x64 tiles per wave

  // ---- encoder layer 0: parallel input GEMM, then sequential recurrence ----
  gemm_bias_nb4_kernel<bf16><<<bigGrid, 128, 0, stream>>>(xbf, I_, wih0b, bias0, Gx0, G4_, I_);
  for (int t = 0; t < T_; ++t) {
    const bf16* hp = t ? H0 + (long)(t - 1) * H_ : zerob;
    long hps = t ? (long)T_ * H_ : (long)H_;
    lstm_step_kernel<<<stepGrid, 128, 0, stream>>>(
        Gx0 + (long)t * G4_, (long)T_ * G4_, hp, hps, whh0b, c0,
        H0 + (long)t * H_, (long)T_ * H_, nullptr, 0, H_);
  }

  // ---- encoder layer 1 ----
  gemm_bias_nb4_kernel<bf16><<<bigGrid, 128, 0, stream>>>(H0, H_, wih1b, bias1, Gx1, G4_, H_);
  for (int t = 0; t < T_; ++t) {
    const bf16* hp = t ? h1p + (size_t)((t - 1) & 1) * B_ * H_ : zerob;
    lstm_step_kernel<<<stepGrid, 128, 0, stream>>>(
        Gx1 + (long)t * G4_, (long)T_ * G4_, hp, (long)H_, whh1b, c1,
        h1p + (size_t)(t & 1) * B_ * H_, (long)H_,
        enc + (long)t * H_, (long)T_ * H_, H_);
  }

  // ---- autoregressive decoder with attention (continues h1/c1 state) ----
  for (int s = 0; s < NA_; ++s) {
    const float* hq  = s ? hdecf : enc + (long)(T_ - 1) * H_;
    long         hqs = s ? (long)H_ : (long)T_ * H_;
    attn_scores_kernel<<<dim3((B_ * T_) / 8), 256, 0, stream>>>(enc, hq, hqs, scores, T_, H_);
    softmax_kernel<<<B_, 256, 0, stream>>>(scores, aw, T_);
    attn_ctx_kernel<<<dim3(H_ / 256, B_), 256, 0, stream>>>(enc, aw, ctx, T_, H_);

    const float* pp = s ? out + (long)(s - 1) * C_ : nullptr;
    xi_kernel<<<(B_ * H_) / 256, 256, 0, stream>>>(ctx, pp, (long)NA_ * C_, xib, H_);

    gemm_bias_kernel<bf16><<<dim3(G4_ / 64, B_ / 16), 128, 0, stream>>>(
        xib, H_, wihdb, biasd, Gxd, G4_, H_);

    const bf16* hp = s ? hdecb : h1p + (size_t)((T_ - 1) & 1) * B_ * H_;
    lstm_step_kernel<<<stepGrid, 128, 0, stream>>>(
        Gxd, (long)G4_, hp, (long)H_, whhdb, c1,
        hdecb, (long)H_, hdecf, (long)H_, H_);

    gemm_bias_kernel<float><<<dim3(C_ / 64, B_ / 16), 128, 0, stream>>>(
        hdecb, H_, wfcb, b_fc, out + (long)s * C_, (long)NA_ * C_, H_);
  }
}
